// CausalSelfAttention_11811160064690
// MI455X (gfx1250) — matrix operand order repaired and checked by reading, not compiled, not run on hardware
//
#include <hip/hip_runtime.h>

// ---------------------------------------------------------------------------
// CDNA5 (gfx1250) bf16 WMMA pipeline for causal GQA attention block.
// GEMMs: TDM (tensor_load_to_lds) double-buffered LDS staging + WMMA.
// Attention: flash-style online softmax with WMMA QK^T and PV.
// ---------------------------------------------------------------------------

typedef __attribute__((ext_vector_type(16))) __bf16       v16bf;
typedef __attribute__((ext_vector_type(8)))  float        v8f;
typedef __attribute__((ext_vector_type(4)))  unsigned int u32x4;
typedef __attribute__((ext_vector_type(8)))  int          i32x8;
typedef __attribute__((ext_vector_type(4)))  int          i32x4;

union BFr { v16bf v; u32x4 u[2]; };

#define WMMA_BF16(A, B, C) \
  __builtin_amdgcn_wmma_f32_16x16x32_bf16(false, (A), false, (B), (short)0, (C), false, false)

static __device__ __forceinline__ float redmax16(float v) {
  v = fmaxf(v, __shfl_xor(v, 1));
  v = fmaxf(v, __shfl_xor(v, 2));
  v = fmaxf(v, __shfl_xor(v, 4));
  v = fmaxf(v, __shfl_xor(v, 8));
  return v;
}
static __device__ __forceinline__ float redsum16(float v) {
  v += __shfl_xor(v, 1);
  v += __shfl_xor(v, 2);
  v += __shfl_xor(v, 4);
  v += __shfl_xor(v, 8);
  return v;
}

// --------------------------- TDM 2D tile -> LDS ----------------------------
// Builds a Tensor DMA Descriptor (D#) per CDNA5 ISA ch.8 and issues
// TENSOR_LOAD_TO_LDS (6-arg builtin on this toolchain). data_size = 2 bytes.
// Padding inserts DWORDs into LDS after each row to kill ds_load bank
// conflicts.

static __device__ __forceinline__ void tdm_load_2d(const void* g, unsigned lds_addr,
                                                   unsigned tile_d0, unsigned tile_d1,
                                                   unsigned stride0,
                                                   unsigned pad_interval,
                                                   unsigned pad_amount) {
  unsigned long long ga = (unsigned long long)g;
  u32x4 g0;
  g0[0] = 1u;                                   // count=1, user descriptor
  g0[1] = lds_addr;                             // LDS byte address
  g0[2] = (unsigned)ga;                         // global_addr[31:0]
  g0[3] = (unsigned)((ga >> 32) & 0x01ffffffu)  // global_addr[56:32]
        | 0x80000000u;                          // type=2 ("image")
  const unsigned td0 = 1u << 30, td1 = 1u << 30;  // huge tensor dims: no OOB
  i32x8 g1;
  g1[0] = (int)((1u << 16)                      // data_size: 2 bytes
              | (1u << 20)                      // pad_enable
              | (pad_interval << 22)            // DWORDs before pad: 2<<code
              | (pad_amount << 25));            // pad DWORDs: code+1
  g1[1] = (int)((td0 & 0xffffu) << 16);         // tensor_dim0[15:0]
  g1[2] = (int)(((td0 >> 16) & 0xffffu) | ((td1 & 0xffffu) << 16));
  g1[3] = (int)(((td1 >> 16) & 0xffffu) | (tile_d0 << 16));
  g1[4] = (int)(tile_d1 & 0xffffu);             // tile_dim1; tile_dim2=0 (2D)
  g1[5] = (int)stride0;                         // tensor_dim0_stride[31:0]
  g1[6] = 0;
  g1[7] = 0;
  i32x4 z4 = {0, 0, 0, 0};                      // groups 2/3: unused dims
  i32x8 z8 = {0, 0, 0, 0, 0, 0, 0, 0};          // trailing words: unused
  __builtin_amdgcn_tensor_load_to_lds(g0, g1, z4, z4, z8, 0);
}

// ------------------------------ conversions --------------------------------

__global__ __launch_bounds__(256) void k_cvt(const float* __restrict__ src,
                                             __bf16* __restrict__ dst, int n) {
  for (int i = blockIdx.x * blockDim.x + threadIdx.x; i < n;
       i += gridDim.x * blockDim.x)
    dst[i] = (__bf16)src[i];
}

// wb[k][n], n<2048: Wq[n][k]; n<2560: Wk[n-2048][k]; else Wv[n-2560][k]
__global__ __launch_bounds__(256) void k_cvt_wcat(const float* __restrict__ Wq,
                                                  const float* __restrict__ Wk,
                                                  const float* __restrict__ Wv,
                                                  __bf16* __restrict__ wb) {
  const int N = 3072, K = 2048, total = K * N;
  for (int i = blockIdx.x * blockDim.x + threadIdx.x; i < total;
       i += gridDim.x * blockDim.x) {
    int k = i / N, n = i % N;
    float v;
    if (n < 2048)      v = Wq[(size_t)n * K + k];
    else if (n < 2560) v = Wk[(size_t)(n - 2048) * K + k];
    else               v = Wv[(size_t)(n - 2560) * K + k];
    wb[i] = (__bf16)v;
  }
}

__global__ __launch_bounds__(256) void k_cvt_wpT(const float* __restrict__ Wp,
                                                 __bf16* __restrict__ wpb) {
  const int N = 2048, K = 2048, total = K * N;
  for (int i = blockIdx.x * blockDim.x + threadIdx.x; i < total;
       i += gridDim.x * blockDim.x) {
    int k = i / N, n = i % N;
    wpb[i] = (__bf16)Wp[(size_t)n * K + k];
  }
}

// --------------------- TDM-staged bf16 WMMA GEMM ---------------------------
// C[M][N] f32 = A[M][K] bf16 (row-major) x B[K][N] bf16 (k-major).
// Block = 8 waves (2 mi x 4 ni) -> 64x256 C tile, BK=32, double-buffered LDS
// filled by the Tensor Data Mover. LDS rows padded +16B by the TDM:
//   A: 64 rows x 80B  (32 bf16 + pad)   -> conflict-free A-operand ds loads
//   B: 32 rows x 528B (256 bf16 + pad)  -> lanes spread across banks

#define LDA_STRIDE 80
#define LDB_STRIDE 528
#define A_BUF_BYTES (64 * LDA_STRIDE)   // 5120
#define B_BUF_BYTES (32 * LDB_STRIDE)   // 16896
#define B_BASE_OFF  (2 * A_BUF_BYTES)   // 10240

__global__ __launch_bounds__(256) void k_gemm_tdm(const __bf16* __restrict__ A,
                                                  const __bf16* __restrict__ B,
                                                  float* __restrict__ C,
                                                  int K, int N) {
  __shared__ alignas(16) unsigned char smem[2 * A_BUF_BYTES + 2 * B_BUF_BYTES];
  const int lane = threadIdx.x & 31;
  const int wave = threadIdx.x >> 5;
  const int lm = lane & 15, g = lane >> 4;
  const int mi = wave & 1;        // 2 M sub-blocks of 32
  const int ni = wave >> 1;       // 4 N sub-blocks of 64
  const int mblk = blockIdx.x * 64;
  const int nblk = blockIdx.y * 256;
  const unsigned lds_base = (unsigned)(unsigned long long)(smem);

  const v8f vz = {0.f, 0.f, 0.f, 0.f, 0.f, 0.f, 0.f, 0.f};
  v8f acc[2][4];
#pragma unroll
  for (int i = 0; i < 2; i++)
#pragma unroll
    for (int s = 0; s < 4; s++) acc[i][s] = vz;

  // prologue: stage k0=0 tiles into buffer 0
  if (threadIdx.x < 32) {
    tdm_load_2d(A + (size_t)mblk * K, lds_base, 32, 64, (unsigned)K, 3u, 3u);
    tdm_load_2d(B + (size_t)nblk, lds_base + B_BASE_OFF, 256, 32, (unsigned)N, 6u, 3u);
  }
  __builtin_amdgcn_s_wait_tensorcnt(0);
  __syncthreads();

  int buf = 0;
  for (int k0 = 0; k0 < K; k0 += 32) {
    const int nxt = k0 + 32;
    if (nxt < K && threadIdx.x < 32) {
      tdm_load_2d(A + (size_t)mblk * K + nxt, lds_base + (buf ^ 1) * A_BUF_BYTES,
                  32, 64, (unsigned)K, 3u, 3u);
      tdm_load_2d(B + (size_t)nxt * N + nblk,
                  lds_base + B_BASE_OFF + (buf ^ 1) * B_BUF_BYTES,
                  256, 32, (unsigned)N, 6u, 3u);
    }
    // compute from current buffer
    const unsigned char* aBase = smem + buf * A_BUF_BYTES;
    const unsigned char* bBase = smem + B_BASE_OFF + buf * B_BUF_BYTES;
    BFr a0, a1;
    {
      const unsigned char* p0 = aBase + (mi * 32 + lm) * LDA_STRIDE + 16 * g;
      const unsigned char* p1 = aBase + (mi * 32 + 16 + lm) * LDA_STRIDE + 16 * g;
      a0.u[0] = *(const u32x4*)p0;
      a0.u[1] = *(const u32x4*)(p0 + 32);
      a1.u[0] = *(const u32x4*)p1;
      a1.u[1] = *(const u32x4*)(p1 + 32);
    }
#pragma unroll
    for (int s = 0; s < 4; s++) {
      BFr b;
      const unsigned char* bp = bBase + ni * 128 + 32 * s + 2 * lm;
#pragma unroll
      for (int e = 0; e < 16; e++)
        b.v[e] = *(const __bf16*)(bp + (e < 8 ? 8 * g + e : 8 + 8 * g + e) * LDB_STRIDE);
      acc[0][s] = WMMA_BF16(a0.v, b.v, acc[0][s]);
      acc[1][s] = WMMA_BF16(a1.v, b.v, acc[1][s]);
    }
    if (nxt < K) {
      __builtin_amdgcn_s_wait_tensorcnt(0);  // all waves drain their TENSORcnt
      __syncthreads();                       // publish buf^1, retire buf reads
      buf ^= 1;
    }
  }

#pragma unroll
  for (int i = 0; i < 2; i++)
#pragma unroll
    for (int s = 0; s < 4; s++)
#pragma unroll
      for (int r = 0; r < 8; r++) {
        int row = mblk + mi * 32 + 16 * i + r + 8 * g;
        int col = nblk + ni * 64 + 16 * s + lm;
        C[(size_t)row * N + col] = acc[i][s][r];
      }
}

// --------------------- RMS-norm + RoPE + layout shuffle --------------------
// grid = (24 slots, T, B); slot<16: q head; 16..19: k head; 20..23: v copy.

__global__ __launch_bounds__(128) void k_normrope(const float* __restrict__ qkv,
                                                  const float* __restrict__ gain,
                                                  __bf16* __restrict__ qb,
                                                  __bf16* __restrict__ kTt,
                                                  __bf16* __restrict__ vb) {
  const int T = 2048;
  const int d = threadIdx.x;
  const int slot = blockIdx.x;
  const int t = blockIdx.y;
  const int b = blockIdx.z;
  __shared__ float sh[128];
  __shared__ float red[128];
  const size_t row = ((size_t)b * T + t) * 3072;

  if (slot >= 20) {  // v: plain bf16 copy into [b][kvh][t][hd]
    int kv = slot - 20;
    float x = qkv[row + 2560 + kv * 128 + d];
    vb[((size_t)(b * 4 + kv) * T + t) * 128 + d] = (__bf16)x;
    return;
  }
  const bool isq = slot < 16;
  const int h = isq ? slot : slot - 16;
  const int off = isq ? h * 128 : 2048 + h * 128;
  float x = qkv[row + off + d];

  red[d] = x * x;
  __syncthreads();
  for (int s = 64; s > 0; s >>= 1) {
    if (d < s) red[d] += red[d + s];
    __syncthreads();
  }
  float inv = rsqrtf(red[0] * (1.f / 128.f) + 1e-6f);
  sh[d] = x * inv;
  __syncthreads();

  float out;
  if (d < 64) {
    int i = d & 31;
    float invf = __expf(-(float)i * 0.28782313662425574f);  // ln(1e4)/32
    float ang = (float)t * invf;
    float c = cosf(ang), sn = sinf(ang);
    if (d < 32) out = sh[d] * c - sh[d + 32] * sn;
    else        out = sh[d] * c + sh[d - 32] * sn;
  } else {
    out = sh[d];
  }
  if (isq) {
    out *= gain[h] * 0.08838834764831845f;  // gain * 1/sqrt(128)
    qb[((size_t)(b * 16 + h) * T + t) * 128 + d] = (__bf16)out;
  } else {
    // transposed K: [b][kvh][hd][t]  (WMMA B-operand friendly)
    kTt[((size_t)(b * 4 + h) * 128 + d) * T + t] = (__bf16)out;
  }
}

// ------------------------- flash attention + ortho -------------------------
// One wave per (b, h, 16-query tile). Online softmax, WMMA for QK^T and PV,
// fused v-orthogonalization epilogue, bf16 token-major output.

__global__ __launch_bounds__(32) void k_attn(const __bf16* __restrict__ qb,
                                             const __bf16* __restrict__ kTt,
                                             const __bf16* __restrict__ vb,
                                             const float* __restrict__ qkv,
                                             __bf16* __restrict__ ybf) {
  const int T = 2048, HD = 128;
  const int qt = blockIdx.x, h = blockIdx.y, b = blockIdx.z;
  const int lane = threadIdx.x, lm = lane & 15, g = lane >> 4;
  const int kvh = h >> 2;
  __shared__ alignas(32) __bf16 pl[16][32];

  BFr qA[4];
  {
    const __bf16* qrow = qb + ((size_t)(b * 16 + h) * T + qt * 16 + lm) * HD;
#pragma unroll
    for (int c = 0; c < 4; c++) {
      qA[c].u[0] = *(const u32x4*)(qrow + 32 * c + 8 * g);
      qA[c].u[1] = *(const u32x4*)(qrow + 32 * c + 16 + 8 * g);
    }
  }

  const v8f vz = {0.f, 0.f, 0.f, 0.f, 0.f, 0.f, 0.f, 0.f};
  v8f acc[8];
  float M[8], L[8];
#pragma unroll
  for (int r = 0; r < 8; r++) { acc[r] = vz; M[r] = -1e30f; L[r] = 0.f; }

  const __bf16* kbase = kTt + (size_t)(b * 4 + kvh) * HD * T;
  const __bf16* vbase = vb + (size_t)(b * 4 + kvh) * T * HD;
  const int lastkey = qt * 16 + 15;

  for (int j0 = 0; j0 <= lastkey; j0 += 32) {
    v8f S[2];
#pragma unroll
    for (int st = 0; st < 2; st++) {
      const int jj = j0 + 16 * st;
      v8f s = vz;
#pragma unroll
      for (int c = 0; c < 4; c++) {
        BFr kb;
        const __bf16* kp = kbase + (size_t)(32 * c) * T + jj + lm;
#pragma unroll
        for (int e = 0; e < 16; e++)
          kb.v[e] = kp[(size_t)(e < 8 ? 8 * g + e : 8 + 8 * g + e) * T];
        s = WMMA_BF16(qA[c].v, kb.v, s);
      }
#pragma unroll
      for (int r = 0; r < 8; r++) {
        int qr = qt * 16 + r + 8 * g;
        int kc = jj + lm;
        if (kc > qr) s[r] = -1e30f;  // causal mask
      }
      S[st] = s;
    }
    float alpha[8];
#pragma unroll
    for (int r = 0; r < 8; r++) {
      float mx = redmax16(fmaxf(S[0][r], S[1][r]));
      float mn = fmaxf(M[r], mx);
      alpha[r] = __expf(M[r] - mn);
      M[r] = mn;
    }
#pragma unroll
    for (int r = 0; r < 8; r++) {
      float p0 = __expf(S[0][r] - M[r]);
      float p1 = __expf(S[1][r] - M[r]);
      S[0][r] = p0;
      S[1][r] = p1;
      L[r] = alpha[r] * L[r] + redsum16(p0 + p1);
    }
#pragma unroll
    for (int ht = 0; ht < 8; ht++)
#pragma unroll
      for (int r = 0; r < 8; r++) acc[ht][r] *= alpha[r];

    __syncthreads();  // protect previous iteration's LDS reads
#pragma unroll
    for (int r = 0; r < 8; r++) {
      pl[r + 8 * g][lm] = (__bf16)S[0][r];
      pl[r + 8 * g][16 + lm] = (__bf16)S[1][r];
    }
    __syncthreads();

    BFr pA;  // re-stripe P (C layout) into WMMA A-operand layout via LDS
    pA.u[0] = *(const u32x4*)&pl[lm][8 * g];
    pA.u[1] = *(const u32x4*)&pl[lm][16 + 8 * g];
#pragma unroll
    for (int ht = 0; ht < 8; ht++) {
      BFr vB;
      const __bf16* vp = vbase + (size_t)j0 * HD + 16 * ht + lm;
#pragma unroll
      for (int e = 0; e < 16; e++)
        vB.v[e] = vp[(size_t)(e < 8 ? 8 * g + e : 8 + 8 * g + e) * HD];
      acc[ht] = WMMA_BF16(pA.v, vB.v, acc[ht]);
    }
  }

  // normalize by softmax denom
#pragma unroll
  for (int r = 0; r < 8; r++) {
    float inv = 1.f / L[r];
#pragma unroll
    for (int ht = 0; ht < 8; ht++) acc[ht][r] *= inv;
  }

  // fused orthogonalization: y -= (y . v_hat) v_hat  (per token, v fp32)
  const float* vsrc = qkv + (size_t)(b * T) * 3072 + 2560 + kvh * 128;
  float dot[8], vns[8];
#pragma unroll
  for (int r = 0; r < 8; r++) { dot[r] = 0.f; vns[r] = 0.f; }
#pragma unroll
  for (int ht = 0; ht < 8; ht++)
#pragma unroll
    for (int r = 0; r < 8; r++) {
      int t = qt * 16 + r + 8 * g;
      float vv = vsrc[(size_t)t * 3072 + 16 * ht + lm];
      dot[r] += acc[ht][r] * vv;
      vns[r] += vv * vv;
    }
  float sc[8];
#pragma unroll
  for (int r = 0; r < 8; r++) {
    float ds = redsum16(dot[r]);
    float ns = redsum16(vns[r]);
    float nrm = fmaxf(sqrtf(ns), 1e-12f);
    sc[r] = ds / (nrm * nrm);
  }
#pragma unroll
  for (int ht = 0; ht < 8; ht++)
#pragma unroll
    for (int r = 0; r < 8; r++) {
      int t = qt * 16 + r + 8 * g;
      float vv = vsrc[(size_t)t * 3072 + 16 * ht + lm];
      float o = acc[ht][r] - sc[r] * vv;
      ybf[((size_t)(b * T + t)) * 2048 + h * 128 + 16 * ht + lm] = (__bf16)o;
    }
}

// ------------------------------- launcher ----------------------------------

extern "C" void kernel_launch(void* const* d_in, const int* in_sizes, int n_in,
                              void* d_out, int out_size, void* d_ws, size_t ws_size,
                              hipStream_t stream) {
  const float* x    = (const float*)d_in[0];
  const float* Wq   = (const float*)d_in[1];
  const float* Wk   = (const float*)d_in[2];
  const float* Wv   = (const float*)d_in[3];
  const float* Wp   = (const float*)d_in[4];
  const float* gain = (const float*)d_in[5];
  float* out = (float*)d_out;
  char* ws = (char*)d_ws;

  // workspace layout (bytes), total ~228 MiB
  __bf16* xb  = (__bf16*)(ws);                 // x bf16            32 MiB
  __bf16* wb  = (__bf16*)(ws + 33554432ull);   // [2048][3072] bf16 12 MiB
  __bf16* wpb = (__bf16*)(ws + 46137344ull);   // [2048][2048] bf16  8 MiB
  float*  qkv = (float*) (ws + 54525952ull);   // [8192][3072] f32  96 MiB
  __bf16* qb  = (__bf16*)(ws + 155189248ull);  // [b][h][t][hd]     32 MiB
  __bf16* kTt = (__bf16*)(ws + 188743680ull);  // [b][kvh][hd][t]    8 MiB
  __bf16* vb  = (__bf16*)(ws + 197132288ull);  // [b][kvh][t][hd]    8 MiB
  __bf16* ybf = (__bf16*)(ws + 205520896ull);  // [8192][2048] bf16 32 MiB

  k_cvt<<<4096, 256, 0, stream>>>(x, xb, 4 * 2048 * 2048);
  k_cvt_wcat<<<3072, 256, 0, stream>>>(Wq, Wk, Wv, wb);
  k_cvt_wpT<<<2048, 256, 0, stream>>>(Wp, wpb);

  // fused QKV projection: [8192 x 2048] x [2048 x 3072], TDM-staged
  k_gemm_tdm<<<dim3(128, 12), 256, 0, stream>>>(xb, wb, qkv, 2048, 3072);

  k_normrope<<<dim3(24, 2048, 4), 128, 0, stream>>>(qkv, gain, qb, kTt, vb);

  // flash attention (+ fused v-orthogonalization)
  k_attn<<<dim3(128, 16, 4), 32, 0, stream>>>(qb, kTt, vb, qkv, ybf);

  // output projection: [8192 x 2048] x [2048 x 2048], TDM-staged
  k_gemm_tdm<<<dim3(128, 8), 256, 0, stream>>>(ybf, wpb, out, 2048, 2048);
}
